// TransMIL_66872640798857
// MI455X (gfx1250) — matrix operand order repaired and checked by reading, not compiled, not run on hardware
//
#include <hip/hip_runtime.h>
#include <cstdint>

#define TPB 256
static constexpr int   HEADS  = 8;
static constexpr int   DH     = 128;
static constexpr int   NPAD   = 4096;   // sequence length incl. cls (pad==0 here)
static constexpr int   DIMC   = 1024;
static constexpr int   MSUB   = 512;    // Nystrom landmarks
static constexpr int   SL     = 32;     // B*HEADS slices
static constexpr float QSCALE = 0.08838834764831845f; // 128^-0.5

typedef __attribute__((ext_vector_type(16))) _Float16 v16h;
typedef __attribute__((ext_vector_type(8)))  float    v8f;
typedef __attribute__((ext_vector_type(4)))  unsigned u32x4;

union FragU { u32x4 q[2]; v16h h; };
union PackU { _Float16 h[2]; unsigned u; };

// ----------------------------------------------------------------------------
// LayerNorm over last dim (one block per row)
// ----------------------------------------------------------------------------
__global__ __launch_bounds__(TPB) void ln_kernel(const float* __restrict__ src,
    float* __restrict__ dst, const float* __restrict__ w, const float* __restrict__ b,
    int D, long srcStride, long dstStride)
{
    long row = blockIdx.x;
    const float* p = src + row * srcStride;
    float* o = dst + row * dstStride;
    __shared__ float red[TPB];
    int tid = threadIdx.x;
    float s = 0.f;
    for (int i = tid; i < D; i += TPB) s += p[i];
    red[tid] = s; __syncthreads();
    for (int st = TPB / 2; st > 0; st >>= 1) { if (tid < st) red[tid] += red[tid + st]; __syncthreads(); }
    float mu = red[0] / (float)D; __syncthreads();
    float v = 0.f;
    for (int i = tid; i < D; i += TPB) { float d = p[i] - mu; v += d * d; }
    red[tid] = v; __syncthreads();
    for (int st = TPB / 2; st > 0; st >>= 1) { if (tid < st) red[tid] += red[tid + st]; __syncthreads(); }
    float inv = rsqrtf(red[0] / (float)D + 1e-5f);
    for (int i = tid; i < D; i += TPB) o[i] = (p[i] - mu) * inv * w[i] + b[i];
}

// ----------------------------------------------------------------------------
// Row softmax (one block per row, in place)
// ----------------------------------------------------------------------------
__global__ __launch_bounds__(TPB) void softmax_kernel(float* __restrict__ x, int L)
{
    long row = blockIdx.x;
    float* p = x + row * (long)L;
    __shared__ float red[TPB];
    int tid = threadIdx.x;
    float mx = -3.4e38f;
    for (int i = tid; i < L; i += TPB) mx = fmaxf(mx, p[i]);
    red[tid] = mx; __syncthreads();
    for (int st = TPB / 2; st > 0; st >>= 1) { if (tid < st) red[tid] = fmaxf(red[tid], red[tid + st]); __syncthreads(); }
    mx = red[0]; __syncthreads();
    float s = 0.f;
    for (int i = tid; i < L; i += TPB) { float e = __expf(p[i] - mx); p[i] = e; s += e; }
    red[tid] = s; __syncthreads();
    for (int st = TPB / 2; st > 0; st >>= 1) { if (tid < st) red[tid] += red[tid + st]; __syncthreads(); }
    float inv = 1.0f / red[0];
    for (int i = tid; i < L; i += TPB) p[i] *= inv;
}

// ----------------------------------------------------------------------------
// Pack A (MxK f32) into WMMA 16x32 f16 fragment tiles, lane-major dwords.
// mode 0: row-major src; mode 1: head-major gather (b,h,pos,dh)->(m,k);
// mode 2: fc1 concat [H(768) | LN(H_kmer)(512)] with M=16380 rows.
// ----------------------------------------------------------------------------
__global__ __launch_bounds__(TPB) void pack_a_kernel(const float* __restrict__ src,
    const float* __restrict__ src2, unsigned* __restrict__ dst,
    int Mt, int Kt, int M, int K, int lda, long sStride, long dStride,
    long tilesPerBatch, int mode, long total)
{
    long t = (long)blockIdx.x * TPB + threadIdx.x;
    if (t >= total) return;
    int  v    = (int)(t & 7);
    int  lane = (int)((t >> 3) & 31);
    long tl   = t >> 8;
    long bt   = tl / tilesPerBatch;
    int  tile = (int)(tl - bt * tilesPerBatch);
    int  mt = tile / Kt, kt = tile - mt * Kt;
    int  hi = lane >> 4;
    int  m  = mt * 16 + (lane & 15);
    int  k0 = kt * 32 + (v < 4 ? hi * 8 + 2 * v : 16 + hi * 8 + 2 * (v - 4));
    float x0 = 0.f, x1 = 0.f;
    if (mode == 0) {
        const float* p = src + bt * sStride;
        if (m < M) {
            if (k0     < K) x0 = p[(long)m * lda + k0];
            if (k0 + 1 < K) x1 = p[(long)m * lda + k0 + 1];
        }
    } else if (mode == 1) { // attnout head-major gather
        if (m < M) {
            int bb = m >> 12, pos = m & (NPAD - 1);
            int h0 = k0 >> 7, d0 = k0 & (DH - 1);
            int k1 = k0 + 1;
            int h1 = k1 >> 7, d1 = k1 & (DH - 1);
            x0 = src[(((long)(bb * HEADS + h0)) * NPAD + pos) * DH + d0];
            x1 = src[(((long)(bb * HEADS + h1)) * NPAD + pos) * DH + d1];
        }
    } else { // fc1 concat
        if (m < M) {
            int bb = m / 4095, pos = m - bb * 4095;
            const float* hr = src + ((long)(bb * 4095 + pos)) * 768;
            x0 = (k0     < 768) ? hr[k0]     : src2[bb * 512 + (k0 - 768)];
            x1 = (k0 + 1 < 768) ? hr[k0 + 1] : src2[bb * 512 + (k0 + 1 - 768)];
        }
    }
    PackU pk; pk.h[0] = (_Float16)x0; pk.h[1] = (_Float16)x1;
    dst[bt * dStride + (long)tile * 256 + lane * 8 + v] = pk.u;
}

// ----------------------------------------------------------------------------
// Pack B (KxN f32) into WMMA 32x16 f16 fragment tiles. trans: src is NxK
// (i.e. we need src^T), reading src[n*ldb + k].
// ----------------------------------------------------------------------------
__global__ __launch_bounds__(TPB) void pack_b_kernel(const float* __restrict__ src,
    unsigned* __restrict__ dst, int Nt, int Kt, int N, int K, int ldb,
    long sStride, long dStride, long tilesPerBatch, int trans, long total)
{
    long t = (long)blockIdx.x * TPB + threadIdx.x;
    if (t >= total) return;
    int  v    = (int)(t & 7);
    int  lane = (int)((t >> 3) & 31);
    long tl   = t >> 8;
    long bt   = tl / tilesPerBatch;
    int  tile = (int)(tl - bt * tilesPerBatch);
    int  nt = tile / Kt, kt = tile - nt * Kt;
    int  n  = nt * 16 + (lane & 15);
    int  k0 = kt * 32 + (v < 4 ? (lane >> 4) * 8 + 2 * v : 16 + (lane >> 4) * 8 + 2 * (v - 4));
    const float* p = src + bt * sStride;
    float x0 = 0.f, x1 = 0.f;
    if (n < N) {
        if (trans) {
            if (k0     < K) x0 = p[(long)n * ldb + k0];
            if (k0 + 1 < K) x1 = p[(long)n * ldb + k0 + 1];
        } else {
            if (k0     < K) x0 = p[(long)k0 * ldb + n];
            if (k0 + 1 < K) x1 = p[(long)(k0 + 1) * ldb + n];
        }
    }
    PackU pk; pk.h[0] = (_Float16)x0; pk.h[1] = (_Float16)x1;
    dst[bt * dStride + (long)tile * 256 + lane * 8 + v] = pk.u;
}

// ----------------------------------------------------------------------------
// Epilogue store of one 16x16 f32 accumulator tile
// ----------------------------------------------------------------------------
__device__ __forceinline__ void store_tile(float* __restrict__ C,
    const float* __restrict__ bias, const float* __restrict__ addsrc, v8f acc,
    int mt, int nt, int M, int N, int ldc, long cBase, float alpha, int mode,
    int relu, int lane)
{
    const int n  = nt * 16 + (lane & 15);
    const int m0 = mt * 16 + ((lane >> 4) << 3);
#pragma unroll
    for (int r = 0; r < 8; ++r) {
        int m = m0 + r;
        if (m >= M || n >= N) continue;
        float val = acc[r] * alpha;
        if (bias) val += bias[n];
        if (relu) val = fmaxf(val, 0.0f);
        if (mode == 0) {
            long idx = cBase + (long)m * ldc + n;
            if (addsrc) val += addsrc[idx];
            C[idx] = val;
        } else if (mode == 1) {
            int bb = m >> 12, pos = m & (NPAD - 1);
            int hh = n >> 7,  dd  = n & (DH - 1);
            C[(((long)(bb * HEADS + hh)) * NPAD + pos) * DH + dd] = val;
        } else {
            int bb = m / 4095, pos = m - bb * 4095;
            C[((long)bb * NPAD + pos + 1) * DIMC + n] = val;
        }
    }
}

// ----------------------------------------------------------------------------
// WMMA GEMM: C = alpha*(A@B) (+bias)(relu)(+addsrc). 8 waves/block, each wave
// owns a 2x2 block of 16x16 f32 tiles (4 accumulators); per K-step it loads
// 2 A frags + 2 B frags and issues 4 v_wmma_f32_16x16x32_f16 (double the
// arithmetic intensity of a 1-tile wave). Block footprint: 64(M) x 128(N).
// mode 0: plain (ldc,cStride); mode 1: qkv head scatter (*alpha = q scale);
// mode 2: fc1 scatter (rows shifted past cls token).
// ----------------------------------------------------------------------------
__global__ __launch_bounds__(TPB) void gemm_wmma_kernel(
    const unsigned* __restrict__ pa, const unsigned* __restrict__ pb,
    float* __restrict__ C, const float* __restrict__ bias, const float* __restrict__ addsrc,
    int Mt, int Nt, int Kt, int M, int N, int ldc,
    long paStride, long pbStride, long cStride, float alpha, int mode, int relu)
{
    const int lane = threadIdx.x & 31;
    const int wv   = threadIdx.x >> 5;
    const int Mt2 = (Mt + 1) >> 1, Nt2 = (Nt + 1) >> 1;
    const int mt2 = blockIdx.y * 2 + (wv >> 2);
    const int nt2 = blockIdx.x * 4 + (wv & 3);
    const int bt = blockIdx.z;
    if (mt2 >= Mt2 || nt2 >= Nt2) return;
    const int  mt0 = mt2 * 2, nt0 = nt2 * 2;
    const bool m1 = (mt0 + 1) < Mt, n1 = (nt0 + 1) < Nt;
    const long tileStride = (long)Kt * 256;

    const unsigned* A0 = pa + (long)bt * paStride + (long)mt0 * tileStride + lane * 8;
    const unsigned* A1 = m1 ? A0 + tileStride : A0;
    const unsigned* B0 = pb + (long)bt * pbStride + (long)nt0 * tileStride + lane * 8;
    const unsigned* B1 = n1 ? B0 + tileStride : B0;

    v8f c00 = {}, c01 = {}, c10 = {}, c11 = {};
    for (int kt = 0; kt < Kt; ++kt) {
        FragU a0, a1, b0, b1;
        a0.q[0] = *(const u32x4*)(A0); a0.q[1] = *(const u32x4*)(A0 + 4);
        a1.q[0] = *(const u32x4*)(A1); a1.q[1] = *(const u32x4*)(A1 + 4);
        b0.q[0] = *(const u32x4*)(B0); b0.q[1] = *(const u32x4*)(B0 + 4);
        b1.q[0] = *(const u32x4*)(B1); b1.q[1] = *(const u32x4*)(B1 + 4);
        if (kt + 2 < Kt) {
            __builtin_prefetch(A0 + 512, 0, 3);
            __builtin_prefetch(A1 + 512, 0, 3);
            __builtin_prefetch(B0 + 512, 0, 3);
            __builtin_prefetch(B1 + 512, 0, 3);
        }
        A0 += 256; A1 += 256; B0 += 256; B1 += 256;
        c00 = __builtin_amdgcn_wmma_f32_16x16x32_f16(false, a0.h, false, b0.h, (short)0, c00, false, false);
        c01 = __builtin_amdgcn_wmma_f32_16x16x32_f16(false, a0.h, false, b1.h, (short)0, c01, false, false);
        c10 = __builtin_amdgcn_wmma_f32_16x16x32_f16(false, a1.h, false, b0.h, (short)0, c10, false, false);
        c11 = __builtin_amdgcn_wmma_f32_16x16x32_f16(false, a1.h, false, b1.h, (short)0, c11, false, false);
    }

    const long cBase = (long)bt * cStride;
    store_tile(C, bias, addsrc, c00, mt0, nt0, M, N, ldc, cBase, alpha, mode, relu, lane);
    if (n1)       store_tile(C, bias, addsrc, c01, mt0,     nt0 + 1, M, N, ldc, cBase, alpha, mode, relu, lane);
    if (m1)       store_tile(C, bias, addsrc, c10, mt0 + 1, nt0,     M, N, ldc, cBase, alpha, mode, relu, lane);
    if (m1 && n1) store_tile(C, bias, addsrc, c11, mt0 + 1, nt0 + 1, M, N, ldc, cBase, alpha, mode, relu, lane);
}

// ----------------------------------------------------------------------------
// k_l / q_l landmark means: mean over groups of 8 positions
// ----------------------------------------------------------------------------
__global__ __launch_bounds__(TPB) void mean8_kernel(const float* __restrict__ in,
    float* __restrict__ out, long total)
{
    long t = (long)blockIdx.x * TPB + threadIdx.x;
    if (t >= total) return;
    long d = t & (DH - 1);
    long m = (t >> 7) & (MSUB - 1);
    long s = t >> 16;                       // MSUB*DH = 65536 per slice
    const float* base = in + ((s * NPAD) + m * 8) * DH + d;
    float acc = 0.f;
#pragma unroll
    for (int j = 0; j < 8; ++j) acc += base[(long)j * DH];
    out[t] = acc * 0.125f;
}

// ----------------------------------------------------------------------------
// pinv scale: pass1 per-slice rowsum/colsum maxima, pass2 global product
// ----------------------------------------------------------------------------
__global__ __launch_bounds__(TPB) void pinv_scale1(const float* __restrict__ a, float* __restrict__ outs)
{
    int s = blockIdx.x;
    const float* p = a + (long)s * MSUB * MSUB;
    __shared__ float red[TPB];
    int tid = threadIdx.x;
    float rmax = 0.f;
    for (int r = tid; r < MSUB; r += TPB) {
        float sum = 0.f;
        for (int j = 0; j < MSUB; ++j) sum += fabsf(p[(long)r * MSUB + j]);
        rmax = fmaxf(rmax, sum);
    }
    red[tid] = rmax; __syncthreads();
    for (int st = TPB / 2; st > 0; st >>= 1) { if (tid < st) red[tid] = fmaxf(red[tid], red[tid + st]); __syncthreads(); }
    if (tid == 0) outs[s * 2] = red[0];
    __syncthreads();
    float cmax = 0.f;
    for (int c = tid; c < MSUB; c += TPB) {
        float sum = 0.f;
        for (int r = 0; r < MSUB; ++r) sum += fabsf(p[(long)r * MSUB + c]);
        cmax = fmaxf(cmax, sum);
    }
    red[tid] = cmax; __syncthreads();
    for (int st = TPB / 2; st > 0; st >>= 1) { if (tid < st) red[tid] = fmaxf(red[tid], red[tid + st]); __syncthreads(); }
    if (tid == 0) outs[s * 2 + 1] = red[0];
}

__global__ void pinv_scale2(float* outs)
{
    __shared__ float r[SL], c[SL];
    int t = threadIdx.x;
    r[t] = outs[t * 2]; c[t] = outs[t * 2 + 1];
    __syncthreads();
    if (t == 0) {
        float rm = 0.f, cm = 0.f;
        for (int i = 0; i < SL; ++i) { rm = fmaxf(rm, r[i]); cm = fmaxf(cm, c[i]); }
        outs[0] = rm * cm;
    }
}

__global__ __launch_bounds__(TPB) void initz_kernel(const float* __restrict__ a,
    const float* __restrict__ scal, float* __restrict__ z, long total)
{
    long t = (long)blockIdx.x * TPB + threadIdx.x;
    if (t >= total) return;
    int  j = (int)(t & (MSUB - 1));
    int  i = (int)((t >> 9) & (MSUB - 1));
    long s = t >> 18;
    z[t] = a[(s << 18) + ((long)j << 9) + i] / scal[0];
}

__global__ __launch_bounds__(TPB) void eyesub_kernel(const float* __restrict__ src,
    float* __restrict__ dst, float c, long total)
{
    long t = (long)blockIdx.x * TPB + threadIdx.x;
    if (t >= total) return;
    int j = (int)(t & (MSUB - 1));
    int i = (int)((t >> 9) & (MSUB - 1));
    float v = -src[t];
    if (i == j) v += c;
    dst[t] = v;
}

// ----------------------------------------------------------------------------
// Depthwise residual conv (k=33) over sequence per head, added into attnout
// ----------------------------------------------------------------------------
__global__ __launch_bounds__(TPB) void resconv_kernel(float* __restrict__ out,
    const float* __restrict__ v, const float* __restrict__ w, long total)
{
    long t = (long)blockIdx.x * TPB + threadIdx.x;
    if (t >= total) return;
    int  d   = (int)(t & (DH - 1));
    long r   = t >> 7;
    int  pos = (int)(r & (NPAD - 1));
    int  s   = (int)(r >> 12);
    int  h   = s & (HEADS - 1);
    float acc = out[t];
    const float* vb = v + ((long)s << 19);  // NPAD*DH = 524288
#pragma unroll
    for (int tt = 0; tt < 33; ++tt) {
        int pp = pos + tt - 16;
        if (pp >= 0 && pp < NPAD) acc += w[h * 33 + tt] * vb[((long)pp << 7) + d];
    }
    out[t] = acc;
}

// ----------------------------------------------------------------------------
// PPEG depthwise conv: dst = src + conv_k(src) + bias on rows 1..4095
// ----------------------------------------------------------------------------
__global__ __launch_bounds__(TPB) void ppeg_kernel(const float* __restrict__ src,
    float* __restrict__ dst, const float* __restrict__ w, const float* __restrict__ bias,
    int kk, long total)
{
    long t = (long)blockIdx.x * TPB + threadIdx.x;
    if (t >= total) return;
    int  c = (int)(t & (DIMC - 1));
    long rr = t >> 10;
    int  r = (int)(rr & (NPAD - 1));
    int  b = (int)(rr >> 12);
    if (r == 0) { dst[t] = src[t]; return; }
    int pos = r - 1;
    int half = kk >> 1;
    float acc = src[t] + bias[c];
    for (int tt = 0; tt < kk; ++tt) {
        int pp = pos + tt - half;
        if (pp >= 0 && pp < NPAD - 1)
            acc += w[c * kk + tt] * src[(((long)b << 12) + (pp + 1)) * DIMC + c];
    }
    dst[t] = acc;
}

__global__ void cls_kernel(const float* __restrict__ cls, float* __restrict__ h)
{
    int t = blockIdx.x * TPB + threadIdx.x;
    if (t >= 4 * DIMC) return;
    int b = t >> 10, c = t & (DIMC - 1);
    h[(long)b * NPAD * DIMC + c] = cls[c];
}

__global__ void fc2_kernel(const float* __restrict__ lnrow, const float* __restrict__ w,
                           const float* __restrict__ b, float* __restrict__ out)
{
    int t = threadIdx.x;
    if (t >= 8) return;
    int bb = t >> 1, c = t & 1;
    float acc = b[c];
    for (int d = 0; d < DIMC; ++d) acc += lnrow[bb * DIMC + d] * w[d * 2 + c];
    out[t] = acc;
}

// ----------------------------------------------------------------------------
// Host-side helpers
// ----------------------------------------------------------------------------
struct WS {
    float *hbuf, *tmp, *attnout, *qb, *kb, *vb;
    float *klb, *qlb, *avb, *pavb, *attn1s, *attn3s;
    float *a2, *zb, *z2b, *azb, *wb;
    float *lnk, *lnrow, *scal;
    unsigned *paPool, *pbPool;
};

static WS carve(void* d_ws)
{
    float* W = (float*)d_ws;
    WS w; size_t off = 0;
    auto take = [&](size_t n) { float* p = W + off; off += n; return p; };
    const size_t H16 = (size_t)4 * NPAD * DIMC;     // 16,777,216
    const size_t SD  = (size_t)SL * MSUB * DH;      // 2,097,152
    const size_t SS  = (size_t)SL * MSUB * MSUB;    // 8,388,608
    w.hbuf = take(H16); w.tmp = take(H16); w.attnout = take(H16);
    w.qb = take(H16);   w.kb = take(H16);  w.vb = take(H16);
    w.klb = take(SD);   w.qlb = take(SD);
    w.avb = take(SD);   w.pavb = take(SD);
    w.attn1s = take((size_t)NPAD * MSUB);
    w.attn3s = take((size_t)MSUB * NPAD);
    w.a2 = take(SS); w.zb = take(SS); w.z2b = take(SS);
    w.azb = take(SS); w.wb = take(SS);
    w.lnk = take(2048); w.lnrow = take(4096); w.scal = take(64);
    w.paPool = (unsigned*)take(16777216);
    w.pbPool = (unsigned*)take(4194304);
    return w;
}

static inline long cdiv(long a, long b) { return (a + b - 1) / b; }

static void launch_pack_a(hipStream_t st, const float* src, const float* src2,
                          unsigned* dst, int M, int K, int lda, int batch,
                          long sStride, int mode)
{
    int Mt = (M + 15) / 16, Kt = (K + 31) / 32;
    long tiles = (long)Mt * Kt, dStride = tiles * 256, total = (long)batch * dStride;
    pack_a_kernel<<<dim3((unsigned)cdiv(total, TPB)), TPB, 0, st>>>(
        src, src2, dst, Mt, Kt, M, K, lda, sStride, dStride, tiles, mode, total);
}

static void launch_pack_b(hipStream_t st, const float* src, unsigned* dst,
                          int N, int K, int ldb, int batch, long sStride, int trans)
{
    int Nt = (N + 15) / 16, Kt = (K + 31) / 32;
    long tiles = (long)Nt * Kt, dStride = tiles * 256, total = (long)batch * dStride;
    pack_b_kernel<<<dim3((unsigned)cdiv(total, TPB)), TPB, 0, st>>>(
        src, dst, Nt, Kt, N, K, ldb, sStride, dStride, tiles, trans, total);
}

static void launch_gemm(hipStream_t st, const unsigned* pa, const unsigned* pb,
                        float* C, const float* bias, const float* add,
                        int M, int N, int K, int ldc, int batch, long cStride,
                        float alpha, int mode, int relu)
{
    int Mt = (M + 15) / 16, Nt = (N + 15) / 16, Kt = (K + 31) / 32;
    long paStride = (long)Mt * Kt * 256, pbStride = (long)Nt * Kt * 256;
    int Mt2 = (Mt + 1) / 2, Nt2 = (Nt + 1) / 2;
    dim3 g((unsigned)((Nt2 + 3) / 4), (unsigned)((Mt2 + 1) / 2), (unsigned)batch);
    gemm_wmma_kernel<<<g, TPB, 0, st>>>(pa, pb, C, bias, add, Mt, Nt, Kt, M, N,
                                        ldc, paStride, pbStride, cStride, alpha, mode, relu);
}

static void attention_layer(hipStream_t st, WS& w,
    const float* ln_w, const float* ln_b, const float* qkv_w,
    const float* out_w, const float* out_b, const float* res_w)
{
    const int HTOT = 4 * NPAD;               // 16384
    const long SS = (long)MSUB * MSUB;
    const long tot512 = (long)SL * SS;

    // LN(h) -> tmp
    ln_kernel<<<(unsigned)HTOT, TPB, 0, st>>>(w.hbuf, w.tmp, ln_w, ln_b, DIMC, (long)DIMC, (long)DIMC);

    // QKV projections (head-major scatter, q gets 1/sqrt(dh))
    launch_pack_a(st, w.tmp, nullptr, w.paPool, HTOT, DIMC, DIMC, 1, 0, 0);
    for (int part = 0; part < 3; ++part) {
        launch_pack_b(st, qkv_w + part * DIMC, w.pbPool, DIMC, DIMC, 3 * DIMC, 1, 0, 0);
        float* dst = part == 0 ? w.qb : (part == 1 ? w.kb : w.vb);
        launch_gemm(st, w.paPool, w.pbPool, dst, nullptr, nullptr,
                    HTOT, DIMC, DIMC, 0, 1, 0, part == 0 ? QSCALE : 1.0f, 1, 0);
    }

    // landmark means
    {
        long tot = (long)SL * MSUB * DH;
        mean8_kernel<<<(unsigned)cdiv(tot, TPB), TPB, 0, st>>>(w.kb, w.klb, tot);
        mean8_kernel<<<(unsigned)cdiv(tot, TPB), TPB, 0, st>>>(w.qb, w.qlb, tot);
    }

    // attn2 = softmax(q_l @ k_l^T), batched over 32 slices
    launch_pack_a(st, w.qlb, nullptr, w.paPool, MSUB, DH, DH, SL, (long)MSUB * DH, 0);
    launch_pack_b(st, w.klb, w.pbPool, MSUB, DH, DH, SL, (long)MSUB * DH, 1);
    launch_gemm(st, w.paPool, w.pbPool, w.a2, nullptr, nullptr,
                MSUB, MSUB, DH, MSUB, SL, SS, 1.0f, 0, 0);
    softmax_kernel<<<SL * MSUB, TPB, 0, st>>>(w.a2, MSUB);

    // Newton-Schulz pseudo-inverse (batched 512^3 WMMA GEMMs)
    pinv_scale1<<<SL, TPB, 0, st>>>(w.a2, w.scal);
    pinv_scale2<<<1, SL, 0, st>>>(w.scal);
    initz_kernel<<<(unsigned)(tot512 / TPB), TPB, 0, st>>>(w.a2, w.scal, w.zb, tot512);

    unsigned* pa_a  = w.paPool;
    unsigned* pa_az = w.paPool + 4194304;
    unsigned* pa_z  = w.paPool + 8388608;
    launch_pack_a(st, w.a2, nullptr, pa_a, MSUB, MSUB, MSUB, SL, SS, 0);
    float* zcur = w.zb; float* znext = w.z2b;
    for (int it = 0; it < 6; ++it) {
        launch_pack_b(st, zcur, w.pbPool, MSUB, MSUB, MSUB, SL, SS, 0);
        launch_gemm(st, pa_a, w.pbPool, w.azb, nullptr, nullptr, MSUB, MSUB, MSUB, MSUB, SL, SS, 1.0f, 0, 0);
        eyesub_kernel<<<(unsigned)(tot512 / TPB), TPB, 0, st>>>(w.azb, w.wb, 7.0f, tot512);   // w1
        launch_pack_a(st, w.azb, nullptr, pa_az, MSUB, MSUB, MSUB, SL, SS, 0);
        launch_pack_b(st, w.wb, w.pbPool, MSUB, MSUB, MSUB, SL, SS, 0);
        launch_gemm(st, pa_az, w.pbPool, znext, nullptr, nullptr, MSUB, MSUB, MSUB, MSUB, SL, SS, 1.0f, 0, 0); // w2
        eyesub_kernel<<<(unsigned)(tot512 / TPB), TPB, 0, st>>>(znext, w.wb, 15.0f, tot512);  // w3
        launch_pack_b(st, w.wb, w.pbPool, MSUB, MSUB, MSUB, SL, SS, 0);
        launch_gemm(st, pa_az, w.pbPool, w.azb, nullptr, nullptr, MSUB, MSUB, MSUB, MSUB, SL, SS, 1.0f, 0, 0); // w4
        eyesub_kernel<<<(unsigned)(tot512 / TPB), TPB, 0, st>>>(w.azb, w.wb, 13.0f, tot512);  // w5
        launch_pack_a(st, zcur, nullptr, pa_z, MSUB, MSUB, MSUB, SL, SS, 0);
        launch_pack_b(st, w.wb, w.pbPool, MSUB, MSUB, MSUB, SL, SS, 0);
        launch_gemm(st, pa_z, w.pbPool, znext, nullptr, nullptr, MSUB, MSUB, MSUB, MSUB, SL, SS, 0.25f, 0, 0);
        float* t = zcur; zcur = znext; znext = t;
    }

    // per-(b,h) slices: attn1, attn3, av, pav, out = attn1 @ (pinv @ (attn3 @ v))
    for (int s = 0; s < SL; ++s) {
        const float* qs  = w.qb  + (long)s * NPAD * DH;
        const float* ks  = w.kb  + (long)s * NPAD * DH;
        const float* vs  = w.vb  + (long)s * NPAD * DH;
        const float* kls = w.klb + (long)s * MSUB * DH;
        const float* qls = w.qlb + (long)s * MSUB * DH;
        // attn1 = softmax(q @ kl^T)
        launch_pack_a(st, qs, nullptr, w.paPool, NPAD, DH, DH, 1, 0, 0);
        launch_pack_b(st, kls, w.pbPool, MSUB, DH, DH, 1, 0, 1);
        launch_gemm(st, w.paPool, w.pbPool, w.attn1s, nullptr, nullptr, NPAD, MSUB, DH, MSUB, 1, 0, 1.0f, 0, 0);
        softmax_kernel<<<NPAD, TPB, 0, st>>>(w.attn1s, MSUB);
        // attn3 = softmax(ql @ k^T)
        launch_pack_a(st, qls, nullptr, w.paPool, MSUB, DH, DH, 1, 0, 0);
        launch_pack_b(st, ks, w.pbPool, NPAD, DH, DH, 1, 0, 1);
        launch_gemm(st, w.paPool, w.pbPool, w.attn3s, nullptr, nullptr, MSUB, NPAD, DH, NPAD, 1, 0, 1.0f, 0, 0);
        softmax_kernel<<<MSUB, TPB, 0, st>>>(w.attn3s, NPAD);
        // av = attn3 @ v
        float* avs = w.avb + (long)s * MSUB * DH;
        launch_pack_a(st, w.attn3s, nullptr, w.paPool, MSUB, NPAD, NPAD, 1, 0, 0);
        launch_pack_b(st, vs, w.pbPool, DH, NPAD, DH, 1, 0, 0);
        launch_gemm(st, w.paPool, w.pbPool, avs, nullptr, nullptr, MSUB, DH, NPAD, DH, 1, 0, 1.0f, 0, 0);
        // pav = z @ av
        float* pavs = w.pavb + (long)s * MSUB * DH;
        launch_pack_a(st, zcur + (long)s * SS, nullptr, w.paPool, MSUB, MSUB, MSUB, 1, 0, 0);
        launch_pack_b(st, avs, w.pbPool, DH, MSUB, DH, 1, 0, 0);
        launch_gemm(st, w.paPool, w.pbPool, pavs, nullptr, nullptr, MSUB, DH, MSUB, DH, 1, 0, 1.0f, 0, 0);
        // out slice = attn1 @ pav
        launch_pack_a(st, w.attn1s, nullptr, w.paPool, NPAD, MSUB, MSUB, 1, 0, 0);
        launch_pack_b(st, pavs, w.pbPool, DH, MSUB, DH, 1, 0, 0);
        launch_gemm(st, w.paPool, w.pbPool, w.attnout + (long)s * NPAD * DH,
                    nullptr, nullptr, NPAD, DH, MSUB, DH, 1, 0, 1.0f, 0, 0);
    }

    // residual depthwise conv on v -> attnout
    {
        long tot = (long)SL * NPAD * DH;
        resconv_kernel<<<(unsigned)(tot / TPB), TPB, 0, st>>>(w.attnout, w.vb, res_w, tot);
    }

    // output projection + bias + residual add into hbuf
    launch_pack_a(st, w.attnout, nullptr, w.paPool, HTOT, DIMC, 0, 1, 0, 1);
    launch_pack_b(st, out_w, w.pbPool, DIMC, DIMC, DIMC, 1, 0, 0);
    launch_gemm(st, w.paPool, w.pbPool, w.hbuf, out_b, w.hbuf, HTOT, DIMC, DIMC, DIMC, 1, 0, 1.0f, 0, 0);
}

// ----------------------------------------------------------------------------
extern "C" void kernel_launch(void* const* d_in, const int* in_sizes, int n_in,
                              void* d_out, int out_size, void* d_ws, size_t ws_size,
                              hipStream_t stream)
{
    (void)in_sizes; (void)n_in; (void)out_size; (void)ws_size;
    const float* H        = (const float*)d_in[0];
    const float* H_kmer   = (const float*)d_in[1];
    const float* nk_w     = (const float*)d_in[2];
    const float* nk_b     = (const float*)d_in[3];
    const float* fc1_w    = (const float*)d_in[4];
    const float* fc1_b    = (const float*)d_in[5];
    const float* cls_tok  = (const float*)d_in[6];
    const float* l1_ln_w  = (const float*)d_in[7];
    const float* l1_ln_b  = (const float*)d_in[8];
    const float* l1_qkv_w = (const float*)d_in[9];
    const float* l1_out_w = (const float*)d_in[10];
    const float* l1_out_b = (const float*)d_in[11];
    const float* l1_res_w = (const float*)d_in[12];
    const float* l2_ln_w  = (const float*)d_in[13];
    const float* l2_ln_b  = (const float*)d_in[14];
    const float* l2_qkv_w = (const float*)d_in[15];
    const float* l2_out_w = (const float*)d_in[16];
    const float* l2_out_b = (const float*)d_in[17];
    const float* l2_res_w = (const float*)d_in[18];
    const float* ppeg_w7  = (const float*)d_in[19];
    const float* ppeg_b7  = (const float*)d_in[20];
    const float* ppeg_w5  = (const float*)d_in[21];
    const float* ppeg_b5  = (const float*)d_in[22];
    const float* ppeg_w3  = (const float*)d_in[23];
    const float* ppeg_b3  = (const float*)d_in[24];
    const float* norm_w   = (const float*)d_in[25];
    const float* norm_b   = (const float*)d_in[26];
    const float* fc2_w    = (const float*)d_in[27];
    const float* fc2_b    = (const float*)d_in[28];

    WS w = carve(d_ws);

    // LN(H_kmer) -> lnk
    ln_kernel<<<4, TPB, 0, stream>>>(H_kmer, w.lnk, nk_w, nk_b, 512, 512L, 512L);

    // fc1: relu(concat(H, lnk) @ fc1_w + b) scattered past the cls row
    launch_pack_a(stream, H, w.lnk, w.paPool, 16380, 1280, 0, 1, 0, 2);
    launch_pack_b(stream, fc1_w, w.pbPool, DIMC, 1280, DIMC, 1, 0, 0);
    launch_gemm(stream, w.paPool, w.pbPool, w.hbuf, fc1_b, nullptr,
                16380, DIMC, 1280, 0, 1, 0, 1.0f, 2, 1);
    cls_kernel<<<16, TPB, 0, stream>>>(cls_tok, w.hbuf);

    // layer 1 attention
    attention_layer(stream, w, l1_ln_w, l1_ln_b, l1_qkv_w, l1_out_w, l1_out_b, l1_res_w);

    // PPEG (three sequential depthwise convs, ping-pong buffers)
    {
        long tot = (long)4 * NPAD * DIMC;
        unsigned g = (unsigned)(tot / TPB);
        ppeg_kernel<<<g, TPB, 0, stream>>>(w.hbuf, w.tmp, ppeg_w7, ppeg_b7, 7, tot);
        ppeg_kernel<<<g, TPB, 0, stream>>>(w.tmp, w.attnout, ppeg_w5, ppeg_b5, 5, tot);
        ppeg_kernel<<<g, TPB, 0, stream>>>(w.attnout, w.hbuf, ppeg_w3, ppeg_b3, 3, tot);
    }

    // layer 2 attention
    attention_layer(stream, w, l2_ln_w, l2_ln_b, l2_qkv_w, l2_out_w, l2_out_b, l2_res_w);

    // final LN of the cls row per batch, then fc2 head
    ln_kernel<<<4, TPB, 0, stream>>>(w.hbuf, w.lnrow, norm_w, norm_b, DIMC,
                                     (long)NPAD * DIMC, (long)DIMC);
    fc2_kernel<<<1, TPB, 0, stream>>>(w.lnrow, fc2_w, fc2_b, (float*)d_out);
}
